// NeuralEpistemicTokenizer_86921548137298
// MI455X (gfx1250) — matrix operand order repaired and checked by reading, not compiled, not run on hardware
//
#include <hip/hip_runtime.h>
#include <hip/hip_bf16.h>

// ---------------- model dims ----------------
#define BB   4
#define TT   2048
#define DD   512
#define HH   8
#define LL   4
#define HIDN 2048
#define DHD  64
#define BT   (BB * TT)
#define NSEG (BB * (TT + 1))
#define EPSF 1e-6f
#define ALPHAF 0.5f

typedef __attribute__((ext_vector_type(16))) __bf16 v16bf;
typedef __attribute__((ext_vector_type(8)))  __bf16 v8bf;
typedef __attribute__((ext_vector_type(8)))  float  v8f;

union U16bf { v16bf v; v8bf h[2]; };

__device__ inline __bf16 to_bf16(float f) {
  union { float f; unsigned u; } x; x.f = f;
  unsigned r = x.u + 0x7FFFu + ((x.u >> 16) & 1u);
  unsigned short s = (unsigned short)(r >> 16);
  union { unsigned short s; __bf16 b; } y; y.s = s;
  return y.b;
}

// ---------------- elementwise kernels ----------------
__global__ void net_convert_bf16(const float* __restrict__ src, __bf16* __restrict__ dst, size_t n) {
  size_t i = (size_t)blockIdx.x * blockDim.x + threadIdx.x;
  if (i < n) dst[i] = to_bf16(src[i]);
}

__global__ void net_embed(const int* __restrict__ tok, const float* __restrict__ E,
                          float* __restrict__ X) {
  size_t i = (size_t)blockIdx.x * blockDim.x + threadIdx.x;
  if (i >= (size_t)BT * DD) return;
  size_t bt = i / DD; int d = (int)(i % DD);
  int t = tok[bt]; t = t < 0 ? 0 : (t > 255 ? 255 : t);
  X[i] = E[(size_t)t * DD + d];
}

__global__ void net_add(float* __restrict__ x, const float* __restrict__ t, size_t n) {
  size_t i = (size_t)blockIdx.x * blockDim.x + threadIdx.x;
  if (i < n) x[i] += t[i];
}

__global__ void net_zero(float* __restrict__ p, size_t n) {
  size_t i = (size_t)blockIdx.x * blockDim.x + threadIdx.x;
  if (i < n) p[i] = 0.0f;
}

__global__ void net_silu_mul(const float* __restrict__ A, const float* __restrict__ B3,
                             __bf16* __restrict__ Y, size_t n) {
  size_t i = (size_t)blockIdx.x * blockDim.x + threadIdx.x;
  if (i >= n) return;
  float a = A[i];
  float s = a / (1.0f + __expf(-a));
  Y[i] = to_bf16(s * B3[i]);
}

// RoPE on fp32 [BT, D] input, writes bf16 [B*H, T, DH]
__global__ void net_rope(const float* __restrict__ QF, __bf16* __restrict__ QB) {
  size_t idx = (size_t)blockIdx.x * blockDim.x + threadIdx.x;
  const size_t total = (size_t)BT * HH * (DHD / 2);
  if (idx >= total) return;
  int i = (int)(idx % (DHD / 2));
  size_t tmp = idx / (DHD / 2);
  int h = (int)(tmp % HH);
  size_t bt = tmp / HH;
  int t = (int)(bt % TT); int b = (int)(bt / TT);
  float x1 = QF[bt * DD + h * DHD + 2 * i];
  float x2 = QF[bt * DD + h * DHD + 2 * i + 1];
  // inv = 10000^(-2i/DH) = exp(-(2i/DH)*ln(10000))
  float inv = __expf(-((float)(2 * i) / (float)DHD) * 9.210340371976184f);
  float ang = (float)t * inv;
  float sn, cs; __sincosf(ang, &sn, &cs);
  size_t base = ((size_t)(b * HH + h) * TT + t) * DHD + 2 * i;
  QB[base]     = to_bf16(x1 * cs - x2 * sn);
  QB[base + 1] = to_bf16(x1 * sn + x2 * cs);
}

// V fp32 [BT, D] -> bf16 transposed [B*H, DH, T]
__global__ void net_vtrans(const float* __restrict__ VF, __bf16* __restrict__ VT) {
  size_t i = (size_t)blockIdx.x * blockDim.x + threadIdx.x;
  if (i >= (size_t)BT * DD) return;
  size_t bt = i / DD; int dcol = (int)(i % DD);
  int h = dcol / DHD, d = dcol % DHD;
  int t = (int)(bt % TT); int b = (int)(bt / TT);
  VT[((size_t)(b * HH + h) * DHD + d) * TT + t] = to_bf16(VF[i]);
}

// ---------------- RMSNorm (one wave per row of D=512) ----------------
__global__ __launch_bounds__(256) void net_rmsnorm_bf16(const float* __restrict__ X,
                                                        const float* __restrict__ Wn,
                                                        __bf16* __restrict__ Y) {
  const int lane = threadIdx.x & 31;
  const int wid  = threadIdx.x >> 5;
  const int row  = blockIdx.x * 8 + wid;
  const float* x = X + (size_t)row * DD;
  float ss = 0.0f;
#pragma unroll
  for (int i = 0; i < DD / 32; ++i) { float v = x[lane + i * 32]; ss += v * v; }
  for (int off = 1; off < 32; off <<= 1) ss += __shfl_xor(ss, off, 32);
  float sc = rsqrtf(ss / (float)DD + EPSF);
  __bf16* y = Y + (size_t)row * DD;
#pragma unroll
  for (int i = 0; i < DD / 32; ++i) { int d = lane + i * 32; y[d] = to_bf16(x[d] * sc * Wn[d]); }
}

__global__ __launch_bounds__(256) void net_rmsnorm_f32(const float* __restrict__ X,
                                                       const float* __restrict__ Wn,
                                                       float* __restrict__ Y) {
  const int lane = threadIdx.x & 31;
  const int wid  = threadIdx.x >> 5;
  const int row  = blockIdx.x * 8 + wid;
  const float* x = X + (size_t)row * DD;
  float ss = 0.0f;
#pragma unroll
  for (int i = 0; i < DD / 32; ++i) { float v = x[lane + i * 32]; ss += v * v; }
  for (int off = 1; off < 32; off <<= 1) ss += __shfl_xor(ss, off, 32);
  float sc = rsqrtf(ss / (float)DD + EPSF);
  float* y = Y + (size_t)row * DD;
#pragma unroll
  for (int i = 0; i < DD / 32; ++i) { int d = lane + i * 32; y[d] = x[d] * sc * Wn[d]; }
}

// ---------------- WMMA GEMM: C[M,N] = A[M,K] * B[N,K]^T ----------------
// A bf16 row-major, Bw bf16 row-major (N rows of K), C fp32 row-major.
// Register-blocked: each wave computes a 32x32 tile (2 A-frags x 2 B-frags ->
// 4 WMMAs per K-step, 2x fragment reuse). Block = 8 waves (2x4) -> 64x128 tile.
__global__ __launch_bounds__(256) void net_gemm_bf16(const __bf16* __restrict__ A,
                                                     const __bf16* __restrict__ Bw,
                                                     float* __restrict__ C,
                                                     int M, int N, int K) {
  const int lane = threadIdx.x & 31;
  const int wid  = threadIdx.x >> 5;
  const int m0 = blockIdx.y * 64  + (wid >> 2) * 32;
  const int n0 = blockIdx.x * 128 + (wid & 3) * 32;
  const int hl = lane >> 4;   // half-wave select
  const int ll = lane & 15;
  v8f acc00 = {}, acc01 = {}, acc10 = {}, acc11 = {};
  // A-frag: row = ll, contiguous K chunk at hl*16
  const __bf16* ap0 = A + (size_t)(m0 + ll) * K + hl * 16;
  const __bf16* ap1 = ap0 + (size_t)16 * K;
  // B-frag: col = ll, contiguous K chunk at hl*16
  const __bf16* bp0 = Bw + (size_t)(n0 + ll) * K + hl * 16;
  const __bf16* bp1 = bp0 + (size_t)16 * K;
  for (int k0 = 0; k0 < K; k0 += 32) {
    U16bf a0, a1, b0, b1;
    a0.h[0] = *(const v8bf*)(ap0 + k0);
    a0.h[1] = *(const v8bf*)(ap0 + k0 + 8);
    a1.h[0] = *(const v8bf*)(ap1 + k0);
    a1.h[1] = *(const v8bf*)(ap1 + k0 + 8);
    b0.v    = *(const v16bf*)(bp0 + k0);
    b1.v    = *(const v16bf*)(bp1 + k0);
    __builtin_prefetch(ap0 + k0 + 128, 0, 1);
    __builtin_prefetch(ap1 + k0 + 128, 0, 1);
    __builtin_prefetch(bp0 + k0 + 128, 0, 1);
    __builtin_prefetch(bp1 + k0 + 128, 0, 1);
    acc00 = __builtin_amdgcn_wmma_f32_16x16x32_bf16(false, a0.v, false, b0.v, (short)0, acc00, false, false);
    acc01 = __builtin_amdgcn_wmma_f32_16x16x32_bf16(false, a0.v, false, b1.v, (short)0, acc01, false, false);
    acc10 = __builtin_amdgcn_wmma_f32_16x16x32_bf16(false, a1.v, false, b0.v, (short)0, acc10, false, false);
    acc11 = __builtin_amdgcn_wmma_f32_16x16x32_bf16(false, a1.v, false, b1.v, (short)0, acc11, false, false);
  }
  float* cp00 = C + (size_t)(m0 + hl * 8) * N + n0 + ll;
#pragma unroll
  for (int r = 0; r < 8; ++r) {
    cp00[(size_t)r * N]             = acc00[r];
    cp00[(size_t)r * N + 16]        = acc01[r];
    cp00[(size_t)(r + 16) * N]      = acc10[r];
    cp00[(size_t)(r + 16) * N + 16] = acc11[r];
  }
}

// ---------------- flash attention: one wave per 16-query tile ----------------
// Q,K bf16 [B*H, T, DH]; Vt bf16 [B*H, DH, T]; O bf16 [B*T, D] (col = h*DH + d)
__global__ __launch_bounds__(128) void net_attn(const __bf16* __restrict__ Q,
                                                const __bf16* __restrict__ Km,
                                                const __bf16* __restrict__ Vt,
                                                __bf16* __restrict__ O) {
  __shared__ __bf16 Pst[4][16 * 32];
  const int lane = threadIdx.x & 31;
  const int wid  = threadIdx.x >> 5;
  const int gw   = blockIdx.x * 4 + wid;
  const int ntq  = TT / 16;
  const int qt = gw % ntq;
  const int bh = gw / ntq;
  const int b = bh / HH, h = bh % HH;
  const int hl = lane >> 4, ll = lane & 15;
  const __bf16* q  = Q  + (size_t)bh * TT * DHD;
  const __bf16* k  = Km + (size_t)bh * TT * DHD;
  const __bf16* vt = Vt + (size_t)bh * DHD * TT;

  U16bf qa[2];
  {
    const __bf16* qp = q + (size_t)(qt * 16 + ll) * DHD + hl * 16;
    qa[0].h[0] = *(const v8bf*)(qp);
    qa[0].h[1] = *(const v8bf*)(qp + 8);
    qa[1].h[0] = *(const v8bf*)(qp + 32);
    qa[1].h[1] = *(const v8bf*)(qp + 40);
  }
  v8f o0 = {}, o1 = {}, o2 = {}, o3 = {};
  float mrow[8], lrow[8];
#pragma unroll
  for (int r = 0; r < 8; ++r) { mrow[r] = -1e30f; lrow[r] = 0.0f; }
  __bf16* P = &Pst[wid][0];

  for (int s0 = 0; s0 < TT; s0 += 32) {
    v8f s0f = {}, s1f = {};
    {
      const __bf16* kp0 = k + (size_t)(s0 + ll) * DHD + hl * 16;
      const __bf16* kp1 = k + (size_t)(s0 + 16 + ll) * DHD + hl * 16;
      U16bf bk;
      bk.v = *(const v16bf*)(kp0);
      s0f = __builtin_amdgcn_wmma_f32_16x16x32_bf16(false, qa[0].v, false, bk.v, (short)0, s0f, false, false);
      bk.v = *(const v16bf*)(kp0 + 32);
      s0f = __builtin_amdgcn_wmma_f32_16x16x32_bf16(false, qa[1].v, false, bk.v, (short)0, s0f, false, false);
      bk.v = *(const v16bf*)(kp1);
      s1f = __builtin_amdgcn_wmma_f32_16x16x32_bf16(false, qa[0].v, false, bk.v, (short)0, s1f, false, false);
      bk.v = *(const v16bf*)(kp1 + 32);
      s1f = __builtin_amdgcn_wmma_f32_16x16x32_bf16(false, qa[1].v, false, bk.v, (short)0, s1f, false, false);
    }
    float fscale[8];
#pragma unroll
    for (int r = 0; r < 8; ++r) {
      float a0 = s0f[r] * 0.125f;   // 1/sqrt(DH)
      float a1 = s1f[r] * 0.125f;
      float mx = fmaxf(a0, a1);
      for (int off = 1; off < 16; off <<= 1) mx = fmaxf(mx, __shfl_xor(mx, off, 32));
      float mn = fmaxf(mrow[r], mx);
      float f  = __expf(mrow[r] - mn);
      float p0 = __expf(a0 - mn), p1 = __expf(a1 - mn);
      float ps = p0 + p1;
      for (int off = 1; off < 16; off <<= 1) ps += __shfl_xor(ps, off, 32);
      lrow[r] = lrow[r] * f + ps;
      mrow[r] = mn;
      fscale[r] = f;
      int row = r + hl * 8;
      P[row * 32 + ll]      = to_bf16(p0);
      P[row * 32 + 16 + ll] = to_bf16(p1);
    }
#pragma unroll
    for (int r = 0; r < 8; ++r) {
      o0[r] *= fscale[r]; o1[r] *= fscale[r]; o2[r] *= fscale[r]; o3[r] *= fscale[r];
    }
    __syncthreads();
    U16bf pa;
    {
      const __bf16* pp = P + ll * 32 + hl * 16;
      pa.h[0] = *(const v8bf*)(pp);
      pa.h[1] = *(const v8bf*)(pp + 8);
    }
    {
      U16bf vb;
      const __bf16* vp = vt + (size_t)ll * TT + s0 + hl * 16;
      vb.v = *(const v16bf*)(vp);
      o0 = __builtin_amdgcn_wmma_f32_16x16x32_bf16(false, pa.v, false, vb.v, (short)0, o0, false, false);
      vb.v = *(const v16bf*)(vp + (size_t)16 * TT);
      o1 = __builtin_amdgcn_wmma_f32_16x16x32_bf16(false, pa.v, false, vb.v, (short)0, o1, false, false);
      vb.v = *(const v16bf*)(vp + (size_t)32 * TT);
      o2 = __builtin_amdgcn_wmma_f32_16x16x32_bf16(false, pa.v, false, vb.v, (short)0, o2, false, false);
      vb.v = *(const v16bf*)(vp + (size_t)48 * TT);
      o3 = __builtin_amdgcn_wmma_f32_16x16x32_bf16(false, pa.v, false, vb.v, (short)0, o3, false, false);
    }
    __syncthreads();
  }
#pragma unroll
  for (int r = 0; r < 8; ++r) {
    float inv = 1.0f / lrow[r];
    int t = qt * 16 + r + hl * 8;
    __bf16* op = O + (size_t)(b * TT + t) * DD + h * DHD + ll;
    op[0]  = to_bf16(o0[r] * inv);
    op[16] = to_bf16(o1[r] * inv);
    op[32] = to_bf16(o2[r] * inv);
    op[48] = to_bf16(o3[r] * inv);
  }
}

// ---------------- segment pooling ----------------
__device__ inline bool net_is_sep(int c) {
  switch (c) {
    case ' ': case '\t': case '\n': case '\r': case '.': case ',': case ';': case ':':
    case '!': case '?': case '(': case ')': case '[': case ']': case '{': case '}':
    case '"': case '\'': case '+': case '-': case '*': case '/': case '=': case '<':
    case '>': case '|': case '&': case '^': case '~': case '%': case '@': case '#':
    case '$': case '\\': return true;
    default: return false;
  }
}

__global__ void net_seg_scan(const int* __restrict__ tok, int* __restrict__ segid) {
  if (threadIdx.x != 0) return;
  int b = blockIdx.x;
  int c = 0;
  for (int t = 0; t < TT; ++t) {
    int v = tok[b * TT + t]; v = v < 0 ? 0 : (v > 255 ? 255 : v);
    c += net_is_sep(v) ? 1 : 0;
    segid[b * TT + t] = c + b * (TT + 1);
  }
}

__global__ void net_seg_accum(const float* __restrict__ XF, const int* __restrict__ segid,
                              float* __restrict__ ssum, float* __restrict__ scnt) {
  int bt = blockIdx.x;
  int s = segid[bt];
  if (threadIdx.x == 0) atomicAdd(&scnt[s], 1.0f);
  for (int d = threadIdx.x; d < DD; d += blockDim.x)
    atomicAdd(&ssum[(size_t)s * DD + d], XF[(size_t)bt * DD + d]);
}

__global__ void net_seg_final(const float* __restrict__ XF, const int* __restrict__ segid,
                              const float* __restrict__ ssum, const float* __restrict__ scnt,
                              float* __restrict__ out) {
  size_t i = (size_t)blockIdx.x * blockDim.x + threadIdx.x;
  if (i >= (size_t)BT * DD) return;
  size_t bt = i / DD; int d = (int)(i % DD);
  int s = segid[bt];
  float cnt = fmaxf(scnt[s], 1.0f);
  out[i] = ALPHAF * XF[i] + (1.0f - ALPHAF) * (ssum[(size_t)s * DD + d] / cnt);
}

// ---------------- host orchestration ----------------
extern "C" void kernel_launch(void* const* d_in, const int* in_sizes, int n_in,
                              void* d_out, int out_size, void* d_ws, size_t ws_size,
                              hipStream_t stream) {
  (void)in_sizes; (void)n_in; (void)out_size; (void)ws_size;
  const int*   tok  = (const int*)d_in[0];
  const float* emb  = (const float*)d_in[1];
  const float* anw  = (const float*)d_in[2];
  const float* wq   = (const float*)d_in[3];
  const float* wk   = (const float*)d_in[4];
  const float* wv   = (const float*)d_in[5];
  const float* wo   = (const float*)d_in[6];
  const float* fnw  = (const float*)d_in[7];
  const float* w1   = (const float*)d_in[8];
  const float* w2   = (const float*)d_in[9];
  const float* w3   = (const float*)d_in[10];
  const float* finw = (const float*)d_in[11];

  char* ws = (char*)d_ws;
  size_t off = 0;
  auto alloc = [&](size_t bytes) -> void* {
    void* p = ws + off;
    off += (bytes + 255) & ~(size_t)255;
    return p;
  };
  const size_t nw4 = (size_t)LL * DD * DD;     // per-array elems for D x D weights
  const size_t nwh = (size_t)LL * HIDN * DD;   // per-array elems for HID x D weights

  __bf16* wqb = (__bf16*)alloc(nw4 * 2);
  __bf16* wkb = (__bf16*)alloc(nw4 * 2);
  __bf16* wvb = (__bf16*)alloc(nw4 * 2);
  __bf16* wob = (__bf16*)alloc(nw4 * 2);
  __bf16* w1b = (__bf16*)alloc(nwh * 2);
  __bf16* w2b = (__bf16*)alloc(nwh * 2);
  __bf16* w3b = (__bf16*)alloc(nwh * 2);
  float*  x   = (float*) alloc((size_t)BT * DD * 4);
  __bf16* xn  = (__bf16*)alloc((size_t)BT * DD * 2);
  float*  h1  = (float*) alloc((size_t)BT * HIDN * 4);
  float*  h3  = (float*) alloc((size_t)BT * HIDN * 4);
  __bf16* hb  = (__bf16*)alloc((size_t)BT * HIDN * 2);
  float*  tmp = (float*) alloc((size_t)BT * DD * 4);
  float*  ssum = (float*)alloc((size_t)NSEG * DD * 4);
  float*  scnt = (float*)alloc((size_t)NSEG * 4);
  int*    segid = (int*) alloc((size_t)BT * 4);

  // aliases (lifetimes disjoint)
  float*  qf = h1;                              // [BT, D]
  float*  vf = h1 + (size_t)BT * DD;            // [BT, D] inside h1 (h1 is 4x larger)
  float*  kf = h3;                              // [BT, D]
  __bf16* qb = hb;                              // [B*H, T, DH]
  __bf16* kb = hb + 1 * (size_t)BT * DD;
  __bf16* vT = hb + 2 * (size_t)BT * DD;        // [B*H, DH, T]
  __bf16* ob = hb + 3 * (size_t)BT * DD;        // [BT, D]
  float*  xf = tmp;                             // final-norm output

  auto cdiv = [](size_t a, size_t b) -> unsigned { return (unsigned)((a + b - 1) / b); };
  const size_t nXD = (size_t)BT * DD;
  const size_t nXH = (size_t)BT * HIDN;

  // weights -> bf16
  net_convert_bf16<<<cdiv(nw4, 256), 256, 0, stream>>>(wq, wqb, nw4);
  net_convert_bf16<<<cdiv(nw4, 256), 256, 0, stream>>>(wk, wkb, nw4);
  net_convert_bf16<<<cdiv(nw4, 256), 256, 0, stream>>>(wv, wvb, nw4);
  net_convert_bf16<<<cdiv(nw4, 256), 256, 0, stream>>>(wo, wob, nw4);
  net_convert_bf16<<<cdiv(nwh, 256), 256, 0, stream>>>(w1, w1b, nwh);
  net_convert_bf16<<<cdiv(nwh, 256), 256, 0, stream>>>(w2, w2b, nwh);
  net_convert_bf16<<<cdiv(nwh, 256), 256, 0, stream>>>(w3, w3b, nwh);

  net_embed<<<cdiv(nXD, 256), 256, 0, stream>>>(tok, emb, x);

  const size_t nrope = (size_t)BT * HH * (DHD / 2);
  for (int l = 0; l < LL; ++l) {
    // ---- attention block ----
    net_rmsnorm_bf16<<<BT / 8, 256, 0, stream>>>(x, anw + (size_t)l * DD, xn);
    net_gemm_bf16<<<dim3(DD / 128, BT / 64), 256, 0, stream>>>(xn, wqb + (size_t)l * DD * DD, qf, BT, DD, DD);
    net_gemm_bf16<<<dim3(DD / 128, BT / 64), 256, 0, stream>>>(xn, wkb + (size_t)l * DD * DD, kf, BT, DD, DD);
    net_gemm_bf16<<<dim3(DD / 128, BT / 64), 256, 0, stream>>>(xn, wvb + (size_t)l * DD * DD, vf, BT, DD, DD);
    net_rope<<<cdiv(nrope, 256), 256, 0, stream>>>(qf, qb);
    net_rope<<<cdiv(nrope, 256), 256, 0, stream>>>(kf, kb);
    net_vtrans<<<cdiv(nXD, 256), 256, 0, stream>>>(vf, vT);
    net_attn<<<(BB * HH * (TT / 16)) / 4, 128, 0, stream>>>(qb, kb, vT, ob);
    net_gemm_bf16<<<dim3(DD / 128, BT / 64), 256, 0, stream>>>(ob, wob + (size_t)l * DD * DD, tmp, BT, DD, DD);
    net_add<<<cdiv(nXD, 256), 256, 0, stream>>>(x, tmp, nXD);

    // ---- FFN block ----
    net_rmsnorm_bf16<<<BT / 8, 256, 0, stream>>>(x, fnw + (size_t)l * DD, xn);
    net_gemm_bf16<<<dim3(HIDN / 128, BT / 64), 256, 0, stream>>>(xn, w1b + (size_t)l * HIDN * DD, h1, BT, HIDN, DD);
    net_gemm_bf16<<<dim3(HIDN / 128, BT / 64), 256, 0, stream>>>(xn, w3b + (size_t)l * HIDN * DD, h3, BT, HIDN, DD);
    net_silu_mul<<<cdiv(nXH, 256), 256, 0, stream>>>(h1, h3, hb, nXH);
    net_gemm_bf16<<<dim3(DD / 128, BT / 64), 256, 0, stream>>>(hb, w2b + (size_t)l * DD * HIDN, tmp, BT, DD, HIDN);
    net_add<<<cdiv(nXD, 256), 256, 0, stream>>>(x, tmp, nXD);
  }

  // ---- final norm + segment pooling ----
  net_rmsnorm_f32<<<BT / 8, 256, 0, stream>>>(x, finw, xf);
  net_seg_scan<<<BB, 32, 0, stream>>>(tok, segid);
  net_zero<<<cdiv((size_t)NSEG * DD, 256), 256, 0, stream>>>(ssum, (size_t)NSEG * DD);
  net_zero<<<cdiv((size_t)NSEG, 256), 256, 0, stream>>>(scnt, (size_t)NSEG);
  net_seg_accum<<<BT, 256, 0, stream>>>(xf, segid, ssum, scnt);
  net_seg_final<<<cdiv(nXD, 256), 256, 0, stream>>>(xf, segid, ssum, scnt, (float*)d_out);
}
